// UltrametricOptimizedTree_74113955659888
// MI455X (gfx1250) — matrix operand order repaired and checked by reading, not compiled, not run on hardware
//
#include <hip/hip_runtime.h>
#include <hip/hip_bf16.h>

typedef __attribute__((ext_vector_type(16))) _Float16 v16h;
typedef __attribute__((ext_vector_type(8)))  float    v8f;

#define NLEAF  4096   // N: leaves (K dimension of the GEMM)
#define MNODE  8191   // M: tree nodes (rows)
#define MPAD   8192   // padded rows (multiple of 32)
#define BDIST  256    // B: batch of distribution pairs (cols)
#define KSTEPS 128    // 4096 / 32 K-steps per WMMA chain

// ---------------------------------------------------------------------------
// Workspace layout (d_ws):
//   [0, 2MiB)            : ddf  — f16 (d1-d2) pre-swizzled into WMMA B-fragments
//   [2MiB, 2MiB+32KiB)   : wv   — edge weights, zero-padded to 8192
//   [2MiB+32KiB, +1KiB)  : w1acc — 256 float partial tree-W1 accumulators
// ---------------------------------------------------------------------------

// dd = d1 - d2, converted to f16 and scattered into per-lane WMMA B-matrix
// fragment layout: B 32x16 f16 tile, lanes 0-15 hold K=0..15 (halfs 0..15),
// lanes 16-31 hold K=16..31 (contiguous K order, same as the A packing).
__global__ void prep_dd(const float* __restrict__ d1, const float* __restrict__ d2,
                        _Float16* __restrict__ ddf) {
  int t = blockIdx.x * blockDim.x + threadIdx.x;  // 0 .. N*B-1  (t = k*B + b)
  int k = t >> 8;          // K index 0..4095
  int b = t & 255;         // column 0..255
  float v = d1[t] - d2[t]; // coalesced: consecutive t -> consecutive b
  int ct   = b >> 4;       // column tile 0..15
  int col  = b & 15;
  int ks   = k >> 5;       // K-step 0..127
  int kin  = k & 31;
  int lane = col + ((kin & 16) ? 16 : 0);
  int h    = kin & 15;
  ddf[(size_t)(((ct << 7) + ks) * 32 + lane) * 16 + h] = (_Float16)v;
}

// wv[m] = param[parents[m]] - param[m]  (0 for the padded row 8191 — this is
// what makes the branch-free row clamp in tw_gemm correct) and zero w1acc
// every call so graph replays stay deterministic.
__global__ void prep_wv(const float* __restrict__ param, const int* __restrict__ parents,
                        float* __restrict__ wv, float* __restrict__ w1acc) {
  int m = blockIdx.x * blockDim.x + threadIdx.x;  // 0..8191
  float w = 0.f;
  if (m < MNODE) w = param[parents[m]] - param[m];
  wv[m] = w;
  if (m < BDIST) w1acc[m] = 0.f;
}

// Fused kernel: C[32 x 256] = subtree_stripe[32 x 4096] @ dd[4096 x 256], then
// w1acc[b] += sum_m wv[m] * |C[m,b]|  (0.5 factor applied in finalize).
// One block per 32-row stripe; 16 wave32s, one 16-col tile each, two 16-row
// WMMA tiles per wave so each B fragment feeds two v_wmma ops.
__global__ __launch_bounds__(512) void tw_gemm(const float* __restrict__ subtree,
                                               const v16h* __restrict__ ddf,
                                               const float* __restrict__ wv,
                                               float* __restrict__ w1acc) {
  // Double-buffered A fragments: 2 bufs x 2 row-tiles x (32 lanes * 16 halfs).
  __shared__ __align__(32) _Float16 aFrag[2][2][512];

  const int tid  = threadIdx.x;
  const int lane = tid & 31;        // lane within wave
  const int ct   = tid >> 5;        // wave id == column tile 0..15
  const int m0   = blockIdx.x << 5; // first row of this 32-row stripe

  // Cooperative A-load role: wave w streams rows w and w+16 (coalesced 128B).
  const int r0  = tid >> 5;         // row-in-stripe 0..15 (tile 0); +16 = tile 1
  const int kin = tid & 31;         // K offset within step
  // WMMA A 16x32 f16 layout (matches ddf B packing):
  //   K 0..15 -> lanes 0-15 halfs 0-15;  K 16..31 -> lanes 16-31 halfs 0-15
  const int laneA = r0 + ((kin & 16) ? 16 : 0);
  const int hA    = kin & 15;

  // Branch-free padding: clamp row index; row 8191 duplicates row 8190 but is
  // multiplied by wv[8191]==0 in the epilogue, so its contribution vanishes.
  int mr0 = m0 + r0;       if (mr0 > MNODE - 1) mr0 = MNODE - 1;
  int mr1 = m0 + r0 + 16;  if (mr1 > MNODE - 1) mr1 = MNODE - 1;
  const float* aPtr0 = subtree + (size_t)mr0 * NLEAF + kin;
  const float* aPtr1 = subtree + (size_t)mr1 * NLEAF + kin;

  v8f acc0 = {};
  v8f acc1 = {};

  // Software pipeline: A-stream values and B fragment fetched one step ahead.
  float av0 = aPtr0[0];
  float av1 = aPtr1[0];
  v16h bcur = ddf[(size_t)((ct << 7) + 0) * 32 + lane];

  for (int ks = 0; ks < KSTEPS; ++ks) {
    const int buf = ks & 1;
    aFrag[buf][0][laneA * 16 + hA] = (_Float16)av0;
    aFrag[buf][1][laneA * 16 + hA] = (_Float16)av1;
    __syncthreads();  // double buffer -> single barrier per step is sufficient

    // Issue next-step loads immediately after the barrier (uniform index,
    // no divergence; last step harmlessly reloads itself).
    const int ksn = (ks + 1 < KSTEPS) ? ks + 1 : ks;
    float av0n = aPtr0[ksn * 32];
    float av1n = aPtr1[ksn * 32];
    v16h  bnxt = ddf[(size_t)((ct << 7) + ksn) * 32 + lane];
    if (ks + 2 < KSTEPS) {                         // uniform scalar branch
      __builtin_prefetch(aPtr0 + (ks + 2) * 32, 0, 1);  // global_prefetch_b8
      __builtin_prefetch(aPtr1 + (ks + 2) * 32, 0, 1);
    }

    v16h a0 = *(const v16h*)&aFrag[buf][0][lane * 16];  // ds_load_b128 x2
    v16h a1 = *(const v16h*)&aFrag[buf][1][lane * 16];
    acc0 = __builtin_amdgcn_wmma_f32_16x16x32_f16(false, a0, false, bcur,
                                                  (short)0, acc0, false, false);
    acc1 = __builtin_amdgcn_wmma_f32_16x16x32_f16(false, a1, false, bcur,
                                                  (short)0, acc1, false, false);
    bcur = bnxt;
    av0  = av0n;
    av1  = av1n;
  }

  // C/D layout: vgpr i, lanes 0-15 -> row base+i, lanes 16-31 -> row base+8+i,
  // column = ct*16 + (lane & 15).  Fuse wv * |c| and reduce rows in-register.
  const int hi = lane >> 4;
  const float* wvp = wv + m0 + (hi << 3);
  float p = 0.f;
#pragma unroll
  for (int i = 0; i < 8; ++i) p += wvp[i]      * __builtin_fabsf(acc0[i]);
#pragma unroll
  for (int i = 0; i < 8; ++i) p += wvp[16 + i] * __builtin_fabsf(acc1[i]);

  // Fold the two half-waves (same column, different row octet).
  p += __shfl_down(p, 16, 32);
  if (lane < 16) atomicAdd(&w1acc[(ct << 4) + lane], p);
}

// out = sum_b (ot[b] - 0.5 * w1[b])^2
__global__ void finalize(const float* __restrict__ ot, const float* __restrict__ w1acc,
                         float* __restrict__ out) {
  __shared__ float red[256];
  int t = threadIdx.x;
  float e = ot[t] - 0.5f * w1acc[t];
  red[t] = e * e;
  __syncthreads();
  for (int s = 128; s > 0; s >>= 1) {
    if (t < s) red[t] += red[t + s];
    __syncthreads();
  }
  if (t == 0) out[0] = red[0];
}

extern "C" void kernel_launch(void* const* d_in, const int* in_sizes, int n_in,
                              void* d_out, int out_size, void* d_ws, size_t ws_size,
                              hipStream_t stream) {
  const float* d1      = (const float*)d_in[0];  // [N, B]
  const float* d2      = (const float*)d_in[1];  // [N, B]
  const float* ot      = (const float*)d_in[2];  // [B]
  const float* subtree = (const float*)d_in[3];  // [M, N]
  const float* param   = (const float*)d_in[4];  // [M]
  const int*   parents = (const int*)d_in[5];    // [M]
  float* out = (float*)d_out;

  char* ws = (char*)d_ws;
  _Float16* ddf   = (_Float16*)ws;                              // 2 MiB
  float*    wv    = (float*)(ws + (size_t)(2u << 20));          // 32 KiB
  float*    w1acc = (float*)(ws + (size_t)(2u << 20) + (32u << 10));

  prep_dd<<<(NLEAF * BDIST) / 256, 256, 0, stream>>>(d1, d2, ddf);
  prep_wv<<<MPAD / 256, 256, 0, stream>>>(param, parents, wv, w1acc);
  tw_gemm<<<MPAD / 32, 512, 0, stream>>>(subtree, (const v16h*)ddf, wv, w1acc);
  finalize<<<1, 256, 0, stream>>>(ot, w1acc, out);
}
